// Model_44375602102863
// MI455X (gfx1250) — matrix operand order repaired and checked by reading, not compiled, not run on hardware
//
#include <hip/hip_runtime.h>

// ---------------- problem constants (match reference) ----------------
#define N_NODES 50000
#define N_EDGES 500000
#define D       128
#define H       4
#define PL      3
#define DK      32
#define NCOL    1152              // 3 mats * 3 rels * 4 heads * 32
#define MT      3128              // ceil(50000/16) row tiles (even)
#define NP      (MT*16)           // 50048 padded rows
#define NT      (NCOL/16)         // 72 col tiles

typedef __attribute__((ext_vector_type(16))) __bf16 v16bf;
typedef __attribute__((ext_vector_type(8)))  float  v8f;
typedef __attribute__((ext_vector_type(4)))  unsigned int u32x4;
typedef __attribute__((ext_vector_type(8)))  int i32x8;
typedef __attribute__((ext_vector_type(4)))  int i32x4;

#if defined(__has_builtin)
#if __has_builtin(__builtin_amdgcn_tensor_load_to_lds) && __has_builtin(__builtin_amdgcn_s_wait_tensorcnt)
#define HAVE_TDM 1
#endif
#endif
#ifndef HAVE_TDM
#define HAVE_TDM 0
#endif

// ---------------- helpers ----------------
__device__ __forceinline__ unsigned short f2bf(float f) {
    unsigned u = __float_as_uint(f);
    unsigned r = u + 0x7FFFu + ((u >> 16) & 1u);   // round-to-nearest-even
    return (unsigned short)(r >> 16);
}
// monotonic float<->uint key for atomicMax on floats
__device__ __forceinline__ unsigned fkey(float f) {
    unsigned u = __float_as_uint(f);
    return (u & 0x80000000u) ? ~u : (u | 0x80000000u);
}
__device__ __forceinline__ float kdec(unsigned k) {
    return __uint_as_float((k & 0x80000000u) ? (k ^ 0x80000000u) : ~k);
}

// ---------------- 0) zero scratch (denom + aggr + headmax, contiguous) ----
__global__ __launch_bounds__(256) void k_init(float* z, long n) {
    long i = (long)blockIdx.x * 256 + threadIdx.x;
    if (i < n) z[i] = 0.0f;
}

// ---------------- 1) LayerNorm -> bf16 row-major (padded rows zeroed) -----
__global__ __launch_bounds__(256) void k_layernorm(const float* __restrict__ x,
                                                   const float* __restrict__ gamma,
                                                   const float* __restrict__ beta,
                                                   unsigned short* __restrict__ xn) {
    int t = threadIdx.x, wave = t >> 5, lane = t & 31;
    long row = (long)blockIdx.x * 8 + wave;
    if (row >= NP) return;
    unsigned short* o = xn + row * D;
    if (row >= N_NODES) {                       // zero pad rows for GEMM
        o[lane] = 0; o[lane + 32] = 0; o[lane + 64] = 0; o[lane + 96] = 0;
        return;
    }
    const float* xr = x + row * D;
    float v0 = xr[lane], v1 = xr[lane + 32], v2 = xr[lane + 64], v3 = xr[lane + 96];
    float s = v0 + v1 + v2 + v3;
    #pragma unroll
    for (int off = 16; off; off >>= 1) s += __shfl_xor(s, off, 32);
    float mu = s * (1.0f / D);
    float d0 = v0 - mu, d1 = v1 - mu, d2 = v2 - mu, d3 = v3 - mu;
    float ss = d0 * d0 + d1 * d1 + d2 * d2 + d3 * d3;
    #pragma unroll
    for (int off = 16; off; off >>= 1) ss += __shfl_xor(ss, off, 32);
    float inv = rsqrtf(ss * (1.0f / D) + 1e-5f);
    o[lane]      = f2bf(d0 * inv * gamma[lane]      + beta[lane]);
    o[lane + 32] = f2bf(d1 * inv * gamma[lane + 32] + beta[lane + 32]);
    o[lane + 64] = f2bf(d2 * inv * gamma[lane + 64] + beta[lane + 64]);
    o[lane + 96] = f2bf(d3 * inv * gamma[lane + 96] + beta[lane + 96]);
}

// ---------------- 2) swizzle activations into WMMA A-operand layout -------
// 16-bit A 16x32: lane l (l<16): row=l, K = {0..7, 16..23}; lane l+16: row=l, K={8..15, 24..31}
__global__ __launch_bounds__(256) void k_packA(const unsigned short* __restrict__ xn,
                                               unsigned short* __restrict__ Asw) {
    int t = threadIdx.x, wave = t >> 5, lane = t & 31;
    int tile = blockIdx.x * 8 + wave;           // [0, MT*4)
    if (tile >= MT * 4) return;
    int mtile = tile >> 2, ks = tile & 3;
    int r = lane & 15, half = lane >> 4;
    const unsigned short* srow = xn + (size_t)(mtile * 16 + r) * D + ks * 32;
    unsigned short* dst = Asw + (size_t)tile * 512 + lane * 16;
    *(uint4*)(dst)     = *(const uint4*)(srow + half * 8);        // K ko..ko+7
    *(uint4*)(dst + 8) = *(const uint4*)(srow + 16 + half * 8);   // K 16+ko..
}

// ---------------- 3) pack weights into WMMA B-operand layout (bf16) -------
// B 32x16 16-bit: lane l: N = l&15, K = {8h..8h+7, 16+8h..16+8h+7}, h = l>>4
__global__ __launch_bounds__(256) void k_packW(const float* __restrict__ wq,
                                               const float* __restrict__ wk,
                                               const float* __restrict__ wv,
                                               const int* __restrict__ nlp,
                                               unsigned short* __restrict__ Bsw) {
    int t = threadIdx.x, wave = t >> 5, lane = t & 31;
    int tile = blockIdx.x * 8 + wave;           // [0, NT*4)
    if (tile >= NT * 4) return;
    int ntile = tile >> 2, ks = tile & 3;
    int layer = *nlp;
    int n  = ntile * 16 + (lane & 15);
    int ko = ks * 32 + ((lane >> 4) << 3);
    int mat = n / 384, rem = n - mat * 384;
    int rel = rem >> 7, rem2 = rem & 127, head = rem2 >> 5, dk = rem2 & 31;
    const float* W = (mat == 0) ? wq : ((mat == 1) ? wk : wv);
    size_t base = ((((size_t)head * PL + layer) * 3 + rel) * D) * DK + dk; // [h][pl][rel][k][dk]
    unsigned short* dst = Bsw + (size_t)tile * 512 + lane * 16;
    #pragma unroll
    for (int j = 0; j < 16; ++j)
        dst[j] = f2bf(W[base + (size_t)(ko + j + (j >= 8 ? 8 : 0)) * DK]);
}

__global__ __launch_bounds__(256) void k_packBias(const float* __restrict__ bq,
                                                  const float* __restrict__ bk,
                                                  const float* __restrict__ bv,
                                                  const int* __restrict__ nlp,
                                                  float* __restrict__ biasP) {
    int n = blockIdx.x * 256 + threadIdx.x;
    if (n >= NCOL) return;
    int layer = *nlp;
    int mat = n / 384, rem = n - mat * 384;
    int rel = rem >> 7, rem2 = rem & 127, head = rem2 >> 5, dk = rem2 & 31;
    const float* B = (mat == 0) ? bq : ((mat == 1) ? bk : bv);
    biasP[n] = B[(((size_t)head * PL + layer) * 3 + rel) * DK + dk];
}

// ---------------- 4) QKV GEMM: xn(50048x128) @ W(128x1152), bf16 WMMA -----
// Block = 2 M-tiles x 8 N-tiles. A (8 KB, pre-swizzled) staged into LDS by the
// Tensor Data Mover (wave 0 issues tensor_load_to_lds, waits TENSORcnt, then
// workgroup barrier). Each wave loads one B fragment per k-step and feeds two
// WMMAs (both M-tiles) -> 8 v_wmma per wave, B traffic halved.
__global__ __launch_bounds__(256) void k_gemm(const unsigned short* __restrict__ Asw,
                                              const unsigned short* __restrict__ Bsw,
                                              const float* __restrict__ biasP,
                                              float* __restrict__ QKV) {
    __shared__ __align__(32) unsigned short As[2 * 4 * 32 * 16];   // 8 KB: two swizzled A tiles
    const int t = threadIdx.x;
    const int mpair = blockIdx.x;                                  // [0, MT/2)
#if HAVE_TDM
    if (t == 0) {
        // ---- build D# (groups 0/1) per CDNA5 ISA ----
        unsigned long long ga = (unsigned long long)(const void*)(Asw + (size_t)mpair * 4096);
        unsigned ldsa = (unsigned)(unsigned long long)(void*)&As[0];  // low 32b = LDS offset
        u32x4 g0;
        g0[0] = 1u;                                            // count=1, user load desc
        g0[1] = ldsa;                                          // lds_addr (bytes)
        g0[2] = (unsigned)(ga & 0xFFFFFFFFull);                // global_addr[31:0]
        g0[3] = (unsigned)((ga >> 32) & 0x01FFFFFFull)         // global_addr[56:32]
              | (2u << 30);                                    // type = 2 ("image")
        const unsigned td0 = 4096, tl0 = 4096, td1 = 1, tl1 = 1, st0 = 4096;
        i32x8 g1;
        g1[0] = (int)(1u << 16);                               // wg_mask=0, data_size=2B
        g1[1] = (int)(td0 << 16);                              // tensor_dim0[15:0] @63:48
        g1[2] = (int)((td0 >> 16) | (td1 << 16));              // dim0[31:16] | dim1[15:0]
        g1[3] = (int)((td1 >> 16) | (tl0 << 16));              // dim1[31:16] | tile_dim0
        g1[4] = (int)(tl1);                                    // tile_dim1 (tile_dim2=0)
        g1[5] = (int)(st0);                                    // tensor_dim0_stride[31:0]
        g1[6] = 0;                                             // stride0 hi | stride1 lo
        g1[7] = 0;
        i32x4 z4 = {0, 0, 0, 0};
        i32x8 z8 = {0, 0, 0, 0, 0, 0, 0, 0};
        // 6-arg form (clang-23 / therock-10.0 headers): groups 0..3 + extra + cpol
        __builtin_amdgcn_tensor_load_to_lds(g0, g1, z4, z4, z8, 0);
        __builtin_amdgcn_s_wait_tensorcnt(0);                  // DMA -> LDS complete
    }
#else
    {   // cooperative fallback: 8 KB = 256 threads x 32 B
        const uint4* g = (const uint4*)(Asw + (size_t)mpair * 4096);
        ((uint4*)As)[t]       = g[t];
        ((uint4*)As)[t + 256] = g[t + 256];
    }
#endif
    __syncthreads();

    const int wave = t >> 5, lane = t & 31;
    const int ntile = blockIdx.y * 8 + wave;
    v8f c0 = {}, c1 = {};
    #pragma unroll
    for (int ks = 0; ks < 4; ++ks) {
        v16bf b  = *(const v16bf*)(Bsw + ((size_t)(ntile * 4 + ks) * 32 + lane) * 16);
        v16bf a0 = *(const v16bf*)&As[(ks * 32 + lane) * 16];
        v16bf a1 = *(const v16bf*)&As[2048 + (ks * 32 + lane) * 16];
        c0 = __builtin_amdgcn_wmma_f32_16x16x32_bf16(false, a0, false, b,
                                                     (short)0, c0, false, false);
        c1 = __builtin_amdgcn_wmma_f32_16x16x32_bf16(false, a1, false, b,
                                                     (short)0, c1, false, false);
    }
    // C/D layout: VGPR j -> row j (lanes 0-15) / row j+8 (lanes 16-31), N = lane&15
    const int ncol  = ntile * 16 + (lane & 15);
    const float bb  = biasP[ncol];
    const int rbase = mpair * 32 + ((lane >> 4) << 3);
    #pragma unroll
    for (int j = 0; j < 8; ++j) {
        QKV[(size_t)(rbase + j)      * NCOL + ncol] = c0[j] + bb;
        QKV[(size_t)(rbase + 16 + j) * NCOL + ncol] = c1[j] + bb;
    }
}

// ---------------- 5) per-edge attention scores + global per-head max ------
__global__ __launch_bounds__(256) void k_edge_attn(const int* __restrict__ ei,
                                                   const float* __restrict__ ev,
                                                   const float* __restrict__ m1,
                                                   const float* __restrict__ m2,
                                                   const float* __restrict__ QKV,
                                                   float* __restrict__ attn,
                                                   unsigned* __restrict__ hmax) {
    __shared__ unsigned smax[H];
    int t = threadIdx.x, wave = t >> 5, lane = t & 31;
    if (t < H) smax[t] = 0u;
    __syncthreads();
    long e = (long)blockIdx.x * 8 + wave;
    if (e < N_EDGES) {
        int src = ei[e], dst = ei[N_EDGES + e];
        int r = (m1[e] > 0.5f) ? 1 : ((m2[e] > 0.5f) ? 2 : 0);
        const float* qb = QKV + (size_t)dst * NCOL + r * 128;          // q block
        const float* kb = QKV + (size_t)src * NCOL + 384 + r * 128;    // k block
        float p0 = qb[lane]      * kb[lane];
        float p1 = qb[lane + 32] * kb[lane + 32];
        float p2 = qb[lane + 64] * kb[lane + 64];
        float p3 = qb[lane + 96] * kb[lane + 96];
        #pragma unroll
        for (int off = 16; off; off >>= 1) {
            p0 += __shfl_xor(p0, off, 32);
            p1 += __shfl_xor(p1, off, 32);
            p2 += __shfl_xor(p2, off, 32);
            p3 += __shfl_xor(p3, off, 32);
        }
        if (lane < H) {
            float dot = (lane == 0) ? p0 : (lane == 1) ? p1 : (lane == 2) ? p2 : p3;
            float dec = expf(-0.105360515657826301f * fabsf(ev[e]));   // 0.9^|t|
            float a   = dot * 0.176776695296636881f * dec;             // /sqrt(32)
            attn[(size_t)lane * N_EDGES + e] = a;
            atomicMax(&smax[lane], fkey(a));
        }
    }
    __syncthreads();
    if (t < H) atomicMax(&hmax[t], smax[t]);
}

// ---------------- 6) exp(attn - max) + segment-sum denominator ------------
__global__ __launch_bounds__(256) void k_edge_exp(const int* __restrict__ ei,
                                                  const unsigned* __restrict__ hmax,
                                                  float* __restrict__ attn,   // in: score, out: ex
                                                  float* __restrict__ denom) {
    long e = (long)blockIdx.x * 256 + threadIdx.x;
    if (e >= N_EDGES) return;
    int dst = ei[N_EDGES + e];
    #pragma unroll
    for (int h = 0; h < H; ++h) {
        float mx = kdec(hmax[h]);
        float ex = expf(attn[(size_t)h * N_EDGES + e] - mx);
        attn[(size_t)h * N_EDGES + e] = ex;
        atomicAdd(&denom[(size_t)h * N_NODES + dst], ex);
    }
}

// ---------------- 7) message aggregation: aggr[dst][h*32+dk] += c * v -----
__global__ __launch_bounds__(256) void k_edge_aggr(const int* __restrict__ ei,
                                                   const float* __restrict__ m1,
                                                   const float* __restrict__ m2,
                                                   const float* __restrict__ QKV,
                                                   const float* __restrict__ ex,
                                                   const float* __restrict__ denom,
                                                   float* __restrict__ aggr) {
    int t = threadIdx.x, wave = t >> 5, lane = t & 31;
    long e = (long)blockIdx.x * 8 + wave;
    if (e >= N_EDGES) return;
    int src = ei[e], dst = ei[N_EDGES + e];
    int r = (m1[e] > 0.5f) ? 1 : ((m2[e] > 0.5f) ? 2 : 0);
    const float* vb = QKV + (size_t)src * NCOL + 768 + r * 128;        // v block
    float* ab = aggr + (size_t)dst * D;
    #pragma unroll
    for (int h = 0; h < H; ++h) {
        float c = ex[(size_t)h * N_EDGES + e] /
                  (denom[(size_t)h * N_NODES + dst] + 1e-16f);
        atomicAdd(&ab[h * 32 + lane], c * vb[h * 32 + lane]);
    }
}

// ---------------- 8) out = x + gelu(aggr) (exact erf GELU, RES=1) ---------
__global__ __launch_bounds__(256) void k_final(const float* __restrict__ x,
                                               const float* __restrict__ aggr,
                                               float* __restrict__ out) {
    long i = (long)blockIdx.x * 256 + threadIdx.x;
    if (i >= (long)N_NODES * D) return;
    float a = aggr[i];
    float g = 0.5f * a * (1.0f + erff(a * 0.70710678118654752f));
    out[i] = x[i] + g;
}

// ---------------- host-side orchestration ----------------
extern "C" void kernel_launch(void* const* d_in, const int* in_sizes, int n_in,
                              void* d_out, int out_size, void* d_ws, size_t ws_size,
                              hipStream_t stream) {
    const float* x     = (const float*)d_in[0];
    const int*   ei    = (const int*)  d_in[1];
    const float* ev    = (const float*)d_in[2];
    const float* m1    = (const float*)d_in[5];
    const float* m2    = (const float*)d_in[6];
    const int*   nlp   = (const int*)  d_in[7];
    const float* w_k   = (const float*)d_in[8];
    const float* b_k   = (const float*)d_in[9];
    const float* w_q   = (const float*)d_in[10];
    const float* b_q   = (const float*)d_in[11];
    const float* w_v   = (const float*)d_in[12];
    const float* b_v   = (const float*)d_in[13];
    const float* gamma = (const float*)d_in[14];
    const float* beta  = (const float*)d_in[15];
    float* out = (float*)d_out;

    // workspace layout (all offsets multiples of 256B)
    char* ws = (char*)d_ws;
    const size_t SZ_XN   = (size_t)NP * D * 2;                 // 12.8 MB bf16 rows
    const size_t SZ_ASW  = (size_t)MT * 4 * 512 * 2;           // 12.8 MB swizzled A
    const size_t SZ_BSW  = (size_t)NT * 4 * 512 * 2;           // 288 KB swizzled B
    const size_t SZ_BIAS = (size_t)NCOL * 4;
    const size_t SZ_QKV  = (size_t)NP * NCOL * 4;              // 230 MB
    const size_t SZ_ATTN = (size_t)H * N_EDGES * 4;            // 8 MB
    const size_t SZ_DEN  = (size_t)H * N_NODES * 4;            // 0.8 MB
    const size_t SZ_AGG  = (size_t)N_NODES * D * 4;            // 25.6 MB

    size_t off = 0;
    unsigned short* xn   = (unsigned short*)(ws + off); off += SZ_XN;
    unsigned short* Asw  = (unsigned short*)(ws + off); off += SZ_ASW;
    unsigned short* Bsw  = (unsigned short*)(ws + off); off += SZ_BSW;
    float*          bias = (float*)(ws + off);          off += ((SZ_BIAS + 255) & ~255ul);
    float*          QKV  = (float*)(ws + off);          off += SZ_QKV;
    float*          attn = (float*)(ws + off);          off += SZ_ATTN;
    float*          den  = (float*)(ws + off);          off += SZ_DEN;   // den, aggr, hmax contiguous
    float*          aggr = (float*)(ws + off);          off += SZ_AGG;
    unsigned*       hmax = (unsigned*)(ws + off);       off += 256;

    // 0) zero denom + aggr + hmax in one pass (contiguous region)
    long nz = (long)(SZ_DEN + SZ_AGG + 256) / 4;
    k_init<<<(nz + 255) / 256, 256, 0, stream>>>(den, nz);
    // 1) LayerNorm -> bf16
    k_layernorm<<<NP / 8, 256, 0, stream>>>(x, gamma, beta, xn);
    // 2) swizzle A
    k_packA<<<(MT * 4 + 7) / 8, 256, 0, stream>>>(xn, Asw);
    // 3) pack weights + bias
    k_packW<<<(NT * 4 + 7) / 8, 256, 0, stream>>>(w_q, w_k, w_v, nlp, Bsw);
    k_packBias<<<(NCOL + 255) / 256, 256, 0, stream>>>(b_q, b_k, b_v, nlp, bias);
    // 4) WMMA GEMM -> QKV table (2 M-tiles x 8 N-tiles per block, TDM A stage)
    k_gemm<<<dim3(MT / 2, NT / 8), 256, 0, stream>>>(Asw, Bsw, bias, QKV);
    // 5) attention scores + per-head global max
    k_edge_attn<<<(N_EDGES + 7) / 8, 256, 0, stream>>>(ei, ev, m1, m2, QKV, attn, hmax);
    // 6) softmax numerator + segment denominator
    k_edge_exp<<<(N_EDGES + 255) / 256, 256, 0, stream>>>(ei, hmax, attn, den);
    // 7) aggregate messages
    k_edge_aggr<<<(N_EDGES + 7) / 8, 256, 0, stream>>>(ei, m1, m2, QKV, attn, den, aggr);
    // 8) residual + GELU
    k_final<<<((long)N_NODES * D + 255) / 256, 256, 0, stream>>>(x, aggr, out);
}
